// MultiHeadSelfAttention_67164698575209
// MI455X (gfx1250) — matrix operand order repaired and checked by reading, not compiled, not run on hardware
//
#include <hip/hip_runtime.h>

// ---------------------------------------------------------------------------
// Multi-head self attention, B=2 S=4096 D=512 H=8 dk=64, fp32 in/out.
// All GEMMs via v_wmma_f32_16x16x32_f16; flash-attention streaming softmax;
// K/V tiles staged into double-buffered LDS by the Tensor Data Mover
// (tensor_load_to_lds + s_wait_tensorcnt), overlapped with WMMA compute.
// Workspace: 4 f16 weights (2MB) + Q,K,Vt,O f16 (32MB) = ~34MB.
// ---------------------------------------------------------------------------

#define D_MODEL 512
#define SEQ     4096
#define BATCH   2
#define HEADS   8
#define DK      64
#define TOK     (BATCH * SEQ)   // 8192 tokens

typedef __attribute__((ext_vector_type(16))) _Float16 v16h;
typedef __attribute__((ext_vector_type(8)))  _Float16 v8h;
typedef __attribute__((ext_vector_type(8)))  float    v8f;
typedef __attribute__((ext_vector_type(4)))  float    f32x4;
typedef __attribute__((ext_vector_type(4)))  unsigned int u32x4;
typedef __attribute__((ext_vector_type(8)))  int      i32x8;
typedef __attribute__((ext_vector_type(4)))  int      i32x4;

// D = A(16x32 f16) * B(32x16 f16) + C(16x16 f32)
static __device__ __forceinline__ v8f wmma16(v16h a, v16h b, v8f c) {
  return __builtin_amdgcn_wmma_f32_16x16x32_f16(false, a, false, b,
                                                (short)0, c, false, false);
}

// A-fragment (16x32, row-major source, f32). p = &src[row*ld + k0],
// kb = (lane>>4)*8. halves 0..15 = K 2*kb..2*kb+15.
static __device__ __forceinline__ v16h a_frag_f32(const float* p, int kb) {
  f32x4 x0 = *(const f32x4*)(p + 2 * kb);
  f32x4 x1 = *(const f32x4*)(p + 2 * kb + 4);
  f32x4 x2 = *(const f32x4*)(p + 2 * kb + 8);
  f32x4 x3 = *(const f32x4*)(p + 2 * kb + 12);
  v16h a;
#pragma unroll
  for (int i = 0; i < 4; ++i) {
    a[i]      = (_Float16)x0[i];
    a[4 + i]  = (_Float16)x1[i];
    a[8 + i]  = (_Float16)x2[i];
    a[12 + i] = (_Float16)x3[i];
  }
  return a;
}

// A-fragment from f16 row-major source (global or LDS).
static __device__ __forceinline__ v16h a_frag_f16(const _Float16* p, int kb) {
  v8h c0 = *(const v8h*)(p + 2 * kb);
  v8h c1 = *(const v8h*)(p + 2 * kb + 8);
  v16h a;
#pragma unroll
  for (int i = 0; i < 8; ++i) { a[i] = c0[i]; a[8 + i] = c1[i]; }
  return a;
}

// ---------------------------------------------------------------------------
// TDM: DMA a 2D tile of 2-byte elements from global into LDS.
// D# layout per CDNA5 ISA:
//   group0: [1:0]=count=1, [63:32]=lds_addr, [120:64]=global_addr, [127:126]=2
//   group1: [17:16]=data_size(1 => 2B), [79:48]=tensor_dim0, [111:80]=tensor_dim1,
//           [127:112]=tile_dim0, [143:128]=tile_dim1, [207:160]=tensor_dim0_stride
//   groups 2/3 (+ extra group on this toolchain): zero — 2D tile, tile_dim2=0,
//   no iterate/gather/pad/multicast.
// This toolchain's builtin is the 6-arg form:
//   (uint32x4 g0, int32x8 g1, int32x4, int32x4, int32x8, i32 cpol)
// ---------------------------------------------------------------------------
static __device__ __forceinline__ void tdm_load_2d(
    unsigned int lds_off, const _Float16* gptr,
    int tensor_d0, int tensor_d1, int tile_d0, int tile_d1, int stride0) {
  unsigned long long ga = (unsigned long long)(uintptr_t)gptr;
  u32x4 g0;
  g0[0] = 1u;                                        // count=1
  g0[1] = lds_off;                                   // LDS byte address
  g0[2] = (unsigned int)ga;                          // global addr [95:64]
  g0[3] = ((unsigned int)(ga >> 32) & 0x01FFFFFFu)   // global addr [120:96]
        | (2u << 30);                                // type=2 ("image")
  i32x8 g1;
  g1[0] = (1 << 16);                                 // data_size=1 (2 bytes)
  g1[1] = (tensor_d0 & 0xFFFF) << 16;                // tensor_dim0 [15:0]
  g1[2] = ((tensor_d0 >> 16) & 0xFFFF) | ((tensor_d1 & 0xFFFF) << 16);
  g1[3] = ((tensor_d1 >> 16) & 0xFFFF) | (tile_d0 << 16);
  g1[4] = tile_d1 & 0xFFFF;                          // tile_dim1, tile_dim2=0
  g1[5] = stride0;                                   // dim0_stride [31:0]
  g1[6] = 0;                                         // dim0_stride hi, dim1_stride lo
  g1[7] = 0;
  i32x4 z4 = {0, 0, 0, 0};
  i32x8 z8 = {0, 0, 0, 0, 0, 0, 0, 0};
  __builtin_amdgcn_tensor_load_to_lds(g0, g1, z4, z4, z8, 0);
}

// ---------------------------------------------------------------------------
// Kernel 1: convert fp32 weights -> f16, transposed to [n][k].
// ---------------------------------------------------------------------------
__global__ __launch_bounds__(256) void wcvt_kernel(
    const float* __restrict__ Wq, const float* __restrict__ Wk,
    const float* __restrict__ Wv, const float* __restrict__ Wo,
    _Float16* __restrict__ wt) {
  const float* W = (blockIdx.y == 0) ? Wq
                 : (blockIdx.y == 1) ? Wk
                 : (blockIdx.y == 2) ? Wv : Wo;
  _Float16* out = wt + (size_t)blockIdx.y * D_MODEL * D_MODEL;
  int i = blockIdx.x * blockDim.x + threadIdx.x;   // out index n*512 + k
  int n = i >> 9, k = i & (D_MODEL - 1);
  out[i] = (_Float16)W[k * D_MODEL + n];
}

// ---------------------------------------------------------------------------
// Kernel 2: fused QKV projection. grid=(TOK/32, 3), block=256 (8 waves).
// Wave computes a 32x64 tile. Q pre-scaled by 0.125; V stored transposed.
// ---------------------------------------------------------------------------
__global__ __launch_bounds__(256) void qkv_kernel(
    const float* __restrict__ x, const _Float16* __restrict__ w16,
    const float* __restrict__ bq, const float* __restrict__ bk,
    const float* __restrict__ bv,
    _Float16* __restrict__ q16, _Float16* __restrict__ k16,
    _Float16* __restrict__ v16t) {
  const int proj = blockIdx.y;
  const _Float16* wt = w16 + (size_t)proj * D_MODEL * D_MODEL;
  const float* bias = (proj == 0) ? bq : (proj == 1) ? bk : bv;
  const int lane = threadIdx.x & 31;
  const int wave = threadIdx.x >> 5;
  const int lo = lane & 15, hi = lane >> 4;
  const int qbase = blockIdx.x * 32;
  const int nb = wave * 64;

  v8f acc[2][4];
#pragma unroll
  for (int rr = 0; rr < 2; ++rr)
#pragma unroll
    for (int cc = 0; cc < 4; ++cc)
#pragma unroll
      for (int r = 0; r < 8; ++r) acc[rr][cc][r] = 0.f;

  for (int kk = 0; kk < D_MODEL; kk += 32) {
    v16h af[2];
#pragma unroll
    for (int rr = 0; rr < 2; ++rr) {
      int m = qbase + rr * 16 + lo;
      af[rr] = a_frag_f32(x + (size_t)m * D_MODEL + kk, hi * 8);
    }
#pragma unroll
    for (int cc = 0; cc < 4; ++cc) {
      int n = nb + cc * 16 + lo;
      v16h bf = *(const v16h*)(wt + (size_t)n * D_MODEL + kk + hi * 16);
#pragma unroll
      for (int rr = 0; rr < 2; ++rr)
        acc[rr][cc] = wmma16(af[rr], bf, acc[rr][cc]);
    }
  }

#pragma unroll
  for (int rr = 0; rr < 2; ++rr) {
#pragma unroll
    for (int cc = 0; cc < 4; ++cc) {
      int n = nb + cc * 16 + lo;
      float bias_n = bias[n];
      int h = n >> 6, d = n & (DK - 1);
#pragma unroll
      for (int r = 0; r < 8; ++r) {
        int t = qbase + rr * 16 + r + hi * 8;
        int b = t >> 12, s = t & (SEQ - 1);
        float val = acc[rr][cc][r] + bias_n;
        if (proj == 0) {
          q16[(((size_t)(b * HEADS + h)) * SEQ + s) * DK + d] =
              (_Float16)(val * 0.125f);
        } else if (proj == 1) {
          k16[(((size_t)(b * HEADS + h)) * SEQ + s) * DK + d] = (_Float16)val;
        } else {
          v16t[(((size_t)(b * HEADS + h)) * DK + d) * SEQ + s] = (_Float16)val;
        }
      }
    }
  }
}

// ---------------------------------------------------------------------------
// Kernel 3: flash attention with TDM-staged K/V. grid=(SEQ/64, B*H),
// block=128 (4 waves). Wave 0 double-buffers 32-key K (32x64) and V (64x32)
// tiles into LDS via tensor_load_to_lds; all waves compute from LDS.
// Per 32-key step: 4 WMMAs scores + online softmax + 4 WMMAs P@V.
// ---------------------------------------------------------------------------
__global__ __launch_bounds__(128) void attn_kernel(
    const _Float16* __restrict__ q16, const _Float16* __restrict__ k16,
    const _Float16* __restrict__ v16t, _Float16* __restrict__ o16) {
  __shared__ __align__(16) _Float16 kbuf[2][32][DK];   // [buf][key][d]
  __shared__ __align__(16) _Float16 vbuf[2][DK][32];   // [buf][d][key]
  __shared__ __align__(16) _Float16 pl[4][16][32];     // per-wave P tile
  const int lane = threadIdx.x & 31;
  const int wave = threadIdx.x >> 5;
  const int lo = lane & 15, hi = lane >> 4;
  const int bh = blockIdx.y;
  const int b = bh >> 3, h = bh & (HEADS - 1);
  const int qbase = blockIdx.x * 64 + wave * 16;
  const _Float16* qp = q16 + (size_t)bh * SEQ * DK;
  const _Float16* kp = k16 + (size_t)bh * SEQ * DK;
  const _Float16* vp = v16t + (size_t)bh * DK * SEQ;

  // Prologue: stage key-block 0 (wave-uniform branch; EXEC stays full).
  if (wave == 0) {
    tdm_load_2d((unsigned int)(uintptr_t)&kbuf[0][0][0], kp,
                DK, SEQ, DK, 32, DK);
    tdm_load_2d((unsigned int)(uintptr_t)&vbuf[0][0][0], vp,
                SEQ, DK, 32, DK, SEQ);
  }

  v16h aq[2];
#pragma unroll
  for (int j = 0; j < 2; ++j)
    aq[j] = a_frag_f16(qp + (size_t)(qbase + lo) * DK + j * 32, hi * 8);

  float mrow[8], lrow[8];
  v8f o[4];
#pragma unroll
  for (int r = 0; r < 8; ++r) { mrow[r] = -3.0e38f; lrow[r] = 0.f; }
#pragma unroll
  for (int cc = 0; cc < 4; ++cc)
#pragma unroll
    for (int r = 0; r < 8; ++r) o[cc][r] = 0.f;

  const float LOG2E = 1.44269504088896340736f;

#pragma unroll 1
  for (int k0 = 0; k0 < SEQ; k0 += 32) {
    const int p = (k0 >> 5) & 1;
    if (wave == 0) __builtin_amdgcn_s_wait_tensorcnt(0);
    __syncthreads();   // buf p ready; everyone done reading buf p^1
    if (wave == 0 && (k0 + 32) < SEQ) {
      tdm_load_2d((unsigned int)(uintptr_t)&kbuf[p ^ 1][0][0],
                  kp + (size_t)(k0 + 32) * DK, DK, SEQ, DK, 32, DK);
      tdm_load_2d((unsigned int)(uintptr_t)&vbuf[p ^ 1][0][0],
                  vp + (k0 + 32), SEQ, DK, 32, DK, SEQ);
    }
    // --- scores: S[16q x 32k], two 16x16 C-tiles, gemm-K = dk = 64 ---
    v8f s[2];
#pragma unroll
    for (int t = 0; t < 2; ++t) {
#pragma unroll
      for (int r = 0; r < 8; ++r) s[t][r] = 0.f;
#pragma unroll
      for (int j = 0; j < 2; ++j) {
        v16h bf = *(const v16h*)(&kbuf[p][t * 16 + lo][j * 32 + hi * 16]);
        s[t] = wmma16(aq[j], bf, s[t]);
      }
    }
    // --- online softmax (row stats replicated across the 16 lanes/row) ---
    float p0[8], p1[8];
#pragma unroll
    for (int r = 0; r < 8; ++r) {
      float mx = fmaxf(s[0][r], s[1][r]);
#pragma unroll
      for (int msk = 1; msk < 16; msk <<= 1)
        mx = fmaxf(mx, __shfl_xor(mx, msk, 32));
      float mnew = fmaxf(mrow[r], mx);
      float scale = __builtin_amdgcn_exp2f((mrow[r] - mnew) * LOG2E);
      mrow[r] = mnew;
      lrow[r] *= scale;
#pragma unroll
      for (int cc = 0; cc < 4; ++cc) o[cc][r] *= scale;
      p0[r] = __builtin_amdgcn_exp2f((s[0][r] - mnew) * LOG2E);
      p1[r] = __builtin_amdgcn_exp2f((s[1][r] - mnew) * LOG2E);
      float rs = p0[r] + p1[r];
#pragma unroll
      for (int msk = 1; msk < 16; msk <<= 1)
        rs += __shfl_xor(rs, msk, 32);
      lrow[r] += rs;
    }
    // --- re-layout P (C layout -> 16x32 row-major in LDS -> A fragment) ---
#pragma unroll
    for (int r = 0; r < 8; ++r) {
      pl[wave][r + hi * 8][lo]      = (_Float16)p0[r];
      pl[wave][r + hi * 8][16 + lo] = (_Float16)p1[r];
    }
    asm volatile("s_wait_dscnt 0" ::: "memory");  // same-wave LDS RAW fence
    v16h ap = a_frag_f16(&pl[wave][lo][0], hi * 8);
    // --- O += P @ V from staged V tile [d][key] ---
#pragma unroll
    for (int cc = 0; cc < 4; ++cc) {
      v16h bf = *(const v16h*)(&vbuf[p][cc * 16 + lo][hi * 16]);
      o[cc] = wmma16(ap, bf, o[cc]);
    }
  }

  // --- normalize, store O as f16 [token][h*64+d] for the final GEMM ---
#pragma unroll
  for (int r = 0; r < 8; ++r) {
    float inv = 1.0f / lrow[r];
    int q = qbase + r + hi * 8;
    size_t base = ((size_t)(b * SEQ + q)) * D_MODEL + h * DK;
#pragma unroll
    for (int cc = 0; cc < 4; ++cc)
      o16[base + cc * 16 + lo] = (_Float16)(o[cc][r] * inv);
  }
}

// ---------------------------------------------------------------------------
// Kernel 4: output projection, fp32 result. grid=(TOK/32), block=256.
// ---------------------------------------------------------------------------
__global__ __launch_bounds__(256) void oproj_kernel(
    const _Float16* __restrict__ o16, const _Float16* __restrict__ wo16t,
    const float* __restrict__ bo, float* __restrict__ out) {
  const int lane = threadIdx.x & 31;
  const int wave = threadIdx.x >> 5;
  const int lo = lane & 15, hi = lane >> 4;
  const int qbase = blockIdx.x * 32;
  const int nb = wave * 64;

  v8f acc[2][4];
#pragma unroll
  for (int rr = 0; rr < 2; ++rr)
#pragma unroll
    for (int cc = 0; cc < 4; ++cc)
#pragma unroll
      for (int r = 0; r < 8; ++r) acc[rr][cc][r] = 0.f;

  for (int kk = 0; kk < D_MODEL; kk += 32) {
    v16h af[2];
#pragma unroll
    for (int rr = 0; rr < 2; ++rr) {
      int m = qbase + rr * 16 + lo;
      af[rr] = a_frag_f16(o16 + (size_t)m * D_MODEL + kk, hi * 8);
    }
#pragma unroll
    for (int cc = 0; cc < 4; ++cc) {
      int n = nb + cc * 16 + lo;
      v16h bf = *(const v16h*)(wo16t + (size_t)n * D_MODEL + kk + hi * 16);
#pragma unroll
      for (int rr = 0; rr < 2; ++rr)
        acc[rr][cc] = wmma16(af[rr], bf, acc[rr][cc]);
    }
  }

#pragma unroll
  for (int rr = 0; rr < 2; ++rr) {
#pragma unroll
    for (int cc = 0; cc < 4; ++cc) {
      int n = nb + cc * 16 + lo;
      float bias_n = bo[n];
#pragma unroll
      for (int r = 0; r < 8; ++r) {
        int t = qbase + rr * 16 + r + hi * 8;
        out[(size_t)t * D_MODEL + n] = acc[rr][cc][r] + bias_n;
      }
    }
  }
}

// ---------------------------------------------------------------------------
extern "C" void kernel_launch(void* const* d_in, const int* in_sizes, int n_in,
                              void* d_out, int out_size, void* d_ws,
                              size_t ws_size, hipStream_t stream) {
  (void)in_sizes; (void)n_in; (void)out_size; (void)ws_size;
  const float* x  = (const float*)d_in[0];
  const float* Wq = (const float*)d_in[1];
  const float* bq = (const float*)d_in[2];
  const float* Wk = (const float*)d_in[3];
  const float* bk = (const float*)d_in[4];
  const float* Wv = (const float*)d_in[5];
  const float* bv = (const float*)d_in[6];
  const float* Wo = (const float*)d_in[7];
  const float* bo = (const float*)d_in[8];
  float* out = (float*)d_out;

  _Float16* ws = (_Float16*)d_ws;
  const size_t WE = (size_t)D_MODEL * D_MODEL;  // 262144 halves per weight
  const size_t QE = (size_t)TOK * D_MODEL;      // 4194304 halves per tensor
  _Float16* w16  = ws;                          // [4][512][512] transposed
  _Float16* q16  = ws + 4 * WE;                 // [B*H][S][dk], pre-scaled
  _Float16* k16  = q16 + QE;                    // [B*H][S][dk]
  _Float16* v16t = k16 + QE;                    // [B*H][dk][S] (transposed)
  _Float16* o16  = v16t + QE;                   // [TOK][512]

  wcvt_kernel<<<dim3((unsigned)(WE / 256), 4), 256, 0, stream>>>(Wq, Wk, Wv,
                                                                 Wo, w16);
  qkv_kernel<<<dim3(TOK / 32, 3), 256, 0, stream>>>(x, w16, bq, bk, bv, q16,
                                                    k16, v16t);
  attn_kernel<<<dim3(SEQ / 64, BATCH * HEADS), 128, 0, stream>>>(q16, k16,
                                                                 v16t, o16);
  oproj_kernel<<<dim3(TOK / 32), 256, 0, stream>>>(o16, w16 + 3 * WE, bo, out);
}
